// GNN_27865747817194
// MI455X (gfx1250) — compile-verified, matrix operand order checked by reading, not run on hardware
//
#include <hip/hip_runtime.h>

// ---------------------------------------------------------------------------
// UniMP TransformerConv GNN forward for MI455X (gfx1250, wave32, WMMA).
// Dense GEMMs -> v_wmma_f32_16x16x32_bf16 (f32 accumulate), LDS-staged A tile
// (async global->LDS). Edge phase -> wave-per-edge, L2-resident gathers +
// atomics.
// ---------------------------------------------------------------------------

#define N_NODES 80000
#define N_EDGES 1280000
#define C 128
#define H 4
#define D 32
#define EPS 1e-5f
#define INV_SQRT_D 0.17677669529663687f

typedef __attribute__((ext_vector_type(16))) __bf16 v16bf;
typedef __attribute__((ext_vector_type(8)))  float  v8f;
typedef __attribute__((ext_vector_type(4)))  int    v4i;

#if __has_builtin(__builtin_amdgcn_global_load_async_to_lds_b128)
#define HAVE_ASYNC_LDS 1
typedef __attribute__((address_space(1))) v4i ga_v4i;   // global int4
typedef __attribute__((address_space(3))) v4i ls_v4i;   // LDS int4
#else
#define HAVE_ASYNC_LDS 0
#endif

__device__ __forceinline__ void wait_async_lds() {
#if HAVE_ASYNC_LDS
#if __has_builtin(__builtin_amdgcn_s_wait_asynccnt)
    __builtin_amdgcn_s_wait_asynccnt(0);
#else
    asm volatile("s_wait_asynccnt 0" ::: "memory");
#endif
#endif
}

// ---------------- utility kernels ----------------

__global__ void k_zero_f32(float* p, int n) {
    int i = blockIdx.x * blockDim.x + threadIdx.x;
    if (i < n) p[i] = 0.0f;
}

__global__ void k_init_max(unsigned* p, int n) {
    int i = blockIdx.x * blockDim.x + threadIdx.x;
    if (i < n) p[i] = 0x007FFFFFu;   // encode(-inf)
}

// Convert & transpose the 18 128x128 weight matrices to bf16 [n][k] layout.
// Slot order: (l*4+m) for m in {q,k,v,s}, then fc1=16, fc2=17.
__global__ void k_cvt_weights(const float* Wq, const float* Wk, const float* Wv,
                              const float* Ws, const float* f1, const float* f2,
                              __bf16* wbf) {
    int gid = blockIdx.x * blockDim.x + threadIdx.x;
    if (gid >= 18 * C * C) return;
    int mat = gid >> 14;          // /16384
    int rem = gid & 16383;
    int n   = rem >> 7;           // output col of original W
    int kk  = rem & 127;          // k index
    const float* src;
    if (mat < 16) {
        int l = mat >> 2, m = mat & 3;
        src = (m == 0 ? Wq : m == 1 ? Wk : m == 2 ? Wv : Ws) + l * C * C;
    } else {
        src = (mat == 16) ? f1 : f2;
    }
    wbf[gid] = (__bf16)src[kk * C + n];   // transposed store
}

// x = concat(static, dynamic, prog, time) @ proj_w   [N,45]@[45,128]
__global__ void k_proj(const float* xs, const float* xd, const float* xp,
                       const float* xt, const float* w, float* x) {
    int gid = blockIdx.x * blockDim.x + threadIdx.x;
    if (gid >= N_NODES * C) return;
    int n = gid >> 7, c = gid & 127;
    float acc = 0.0f;
    #pragma unroll
    for (int k = 0; k < 22; ++k) acc += xs[n * 22 + k] * w[k * C + c];
    #pragma unroll
    for (int k = 0; k < 12; ++k) acc += xd[n * 12 + k] * w[(22 + k) * C + c];
    #pragma unroll
    for (int k = 0; k < 7; ++k)  acc += xp[n * 7 + k]  * w[(34 + k) * C + c];
    #pragma unroll
    for (int k = 0; k < 4; ++k)  acc += xt[n * 4 + k]  * w[(41 + k) * C + c];
    x[gid] = acc;
}

// LayerNorm, one wave32 per node; lane handles float4 of channels.
__global__ void k_layernorm(const float* x, const float* g, const float* b,
                            float* h) {
    int wid  = (blockIdx.x * blockDim.x + threadIdx.x) >> 5;
    int lane = threadIdx.x & 31;
    if (wid >= N_NODES) return;
    float4 v = ((const float4*)(x + (size_t)wid * C))[lane];
    float s = v.x + v.y + v.z + v.w;
    #pragma unroll
    for (int off = 16; off >= 1; off >>= 1) s += __shfl_xor(s, off, 32);
    float mu = s * (1.0f / C);
    float dx = v.x - mu, dy = v.y - mu, dz = v.z - mu, dw = v.w - mu;
    float q = dx * dx + dy * dy + dz * dz + dw * dw;
    #pragma unroll
    for (int off = 16; off >= 1; off >>= 1) q += __shfl_xor(q, off, 32);
    float rs = rsqrtf(q * (1.0f / C) + EPS);
    int c0 = lane * 4;
    float4 o;
    o.x = dx * rs * g[c0 + 0] + b[c0 + 0];
    o.y = dy * rs * g[c0 + 1] + b[c0 + 1];
    o.z = dz * rs * g[c0 + 2] + b[c0 + 2];
    o.w = dw * rs * g[c0 + 3] + b[c0 + 3];
    ((float4*)(h + (size_t)wid * C))[lane] = o;
}

// ---------------- WMMA GEMM helpers -----------------------------------------
// A fragment (ISA 7.12.2, 16-bit A 16x32): lane<16 -> M=lane, K-pairs
// {0,1},{2,3},{4,5},{6,7},{16..23}; lanes 16-31 same rows, K+8 group.
// C/D f32 16x16: lane<16 -> N=lane, VGPR j -> M=j; lanes 16-31 -> M=j+8.

__device__ __forceinline__ v16bf load_b_frag(const __bf16* __restrict__ brow,
                                             int kbase) {
    v16bf b;
    #pragma unroll
    for (int i = 0; i < 8; ++i) {
        int k0 = kbase + ((i < 4) ? 0 : 16) + 2 * (i & 3);
        b[2 * i]     = brow[k0];
        b[2 * i + 1] = brow[k0 + 1];
    }
    return b;
}

// Fused q/k/v/s GEMM: [16 rows,128] @ 4x[128,128]bf16 + bias.
// 256 threads = 8 waves. A tile staged once in LDS (async when available),
// decoded once per wave into 4 k-step fragments, reused for all 4 matrices.
__global__ void __launch_bounds__(256)
k_gemm_qkvs(const float* __restrict__ act, const __bf16* __restrict__ w4,
            const float* __restrict__ bq, const float* __restrict__ bk,
            const float* __restrict__ bv, const float* __restrict__ bs,
            float* __restrict__ oq, float* __restrict__ ok,
            float* __restrict__ ov, float* __restrict__ os) {
    __shared__ float Atile[16 * C];          // 8 KB
    int tid  = threadIdx.x;
    int wave = tid >> 5;
    int lane = tid & 31;
    int r    = lane & 15;
    int half = lane >> 4;
    int rowbase = blockIdx.x * 16;

    // ---- stage A tile: 16x128 f32 = 512 float4 chunks, 2 per thread ----
    const float* gsrc = act + (size_t)rowbase * C;
#if HAVE_ASYNC_LDS
    __builtin_amdgcn_global_load_async_to_lds_b128(
        (ga_v4i*)(gsrc + tid * 4), (ls_v4i*)&Atile[tid * 4], 0, 0);
    __builtin_amdgcn_global_load_async_to_lds_b128(
        (ga_v4i*)(gsrc + 1024 + tid * 4), (ls_v4i*)&Atile[1024 + tid * 4], 0, 0);
    wait_async_lds();
#else
    ((float4*)Atile)[tid]       = ((const float4*)gsrc)[tid];
    ((float4*)Atile)[tid + 256] = ((const float4*)gsrc)[tid + 256];
#endif
    __syncthreads();

    // ---- decode A fragments for all 4 k-steps (reused across 4 matrices) ----
    v16bf a[4];
    const float* arow = &Atile[r * C];
    #pragma unroll
    for (int kk = 0; kk < 4; ++kk) {
        int kbase = kk * 32 + half * 8;
        #pragma unroll
        for (int i = 0; i < 8; ++i) {
            int k0 = kbase + ((i < 4) ? 0 : 16) + 2 * (i & 3);
            float2 af = *(const float2*)(arow + k0);
            a[kk][2 * i]     = (__bf16)af.x;
            a[kk][2 * i + 1] = (__bf16)af.y;
        }
    }

    int colbase = wave * 16;
    int col = colbase + r;
    #pragma unroll
    for (int m = 0; m < 4; ++m) {
        const __bf16* brow = w4 + (size_t)m * C * C + (size_t)(colbase + r) * C;
        v8f acc = {};
        #pragma unroll
        for (int kk = 0; kk < 4; ++kk) {
            v16bf bfr = load_b_frag(brow, kk * 32 + half * 8);
            acc = __builtin_amdgcn_wmma_f32_16x16x32_bf16(
                false, a[kk], false, bfr, (short)0, acc, false, false);
        }
        const float* bias = (m == 0) ? bq : (m == 1) ? bk : (m == 2) ? bv : bs;
        float* out        = (m == 0) ? oq : (m == 1) ? ok : (m == 2) ? ov : os;
        float bval = bias[col];
        #pragma unroll
        for (int j = 0; j < 8; ++j) {
            int mm = rowbase + j + half * 8;
            out[(size_t)mm * C + col] = acc[j] + bval;
        }
    }
}

// Single-matrix GEMM (fc1/fc2) with leaky_relu(0.2) epilogue.
__global__ void __launch_bounds__(256)
k_gemm128(const float* __restrict__ act, const __bf16* __restrict__ wT,
          const float* __restrict__ bias, float* __restrict__ out) {
    int wave = threadIdx.x >> 5;
    int lane = threadIdx.x & 31;
    int r    = lane & 15;
    int half = lane >> 4;
    int rowbase = blockIdx.x * 16;
    int colbase = wave * 16;
    const float*  arow = act + (size_t)(rowbase + r) * C;
    const __bf16* brow = wT  + (size_t)(colbase + r) * C;
    v8f acc = {};
    #pragma unroll
    for (int kk = 0; kk < 4; ++kk) {
        int kbase = kk * 32 + half * 8;
        v16bf a;
        #pragma unroll
        for (int i = 0; i < 8; ++i) {
            int k0 = kbase + ((i < 4) ? 0 : 16) + 2 * (i & 3);
            float2 af = *(const float2*)(arow + k0);
            a[2 * i]     = (__bf16)af.x;
            a[2 * i + 1] = (__bf16)af.y;
        }
        v16bf bfr = load_b_frag(brow, kbase);
        acc = __builtin_amdgcn_wmma_f32_16x16x32_bf16(
            false, a, false, bfr, (short)0, acc, false, false);
    }
    int col = colbase + r;
    float bv = bias[col];
    #pragma unroll
    for (int j = 0; j < 8; ++j) {
        int m = rowbase + j + half * 8;
        float v = acc[j] + bv;
        out[(size_t)m * C + col] = (v > 0.0f) ? v : 0.2f * v;
    }
}

// ---------------- edge phase ----------------

__device__ __forceinline__ unsigned enc_f32(float f) {
    unsigned u = __float_as_uint(f);
    return (u & 0x80000000u) ? ~u : (u | 0x80000000u);
}

// One wave per edge: lane L owns channels {h*32+L}. Four head dot products
// via per-lane products + butterfly reduction. atomicMax segment max.
__global__ void k_edge_scores(const int* __restrict__ ei,
                              const float* __restrict__ q,
                              const float* __restrict__ k,
                              float* __restrict__ score,
                              unsigned* __restrict__ mx) {
    int wid  = (blockIdx.x * blockDim.x + threadIdx.x) >> 5;
    int lane = threadIdx.x & 31;
    if (wid >= N_EDGES) return;
    if (lane == 0 && wid + 4096 < N_EDGES) {
        __builtin_prefetch(ei + wid + 4096, 0, 0);             // stream src idx
        __builtin_prefetch(ei + N_EDGES + wid + 4096, 0, 0);   // stream dst idx
    }
    int s = ei[wid], d = ei[N_EDGES + wid];
    const float* qd = q + (size_t)d * C;
    const float* ks = k + (size_t)s * C;
    float p0 = qd[lane]      * ks[lane];
    float p1 = qd[32 + lane] * ks[32 + lane];
    float p2 = qd[64 + lane] * ks[64 + lane];
    float p3 = qd[96 + lane] * ks[96 + lane];
    #pragma unroll
    for (int off = 16; off >= 1; off >>= 1) {
        p0 += __shfl_xor(p0, off, 32);
        p1 += __shfl_xor(p1, off, 32);
        p2 += __shfl_xor(p2, off, 32);
        p3 += __shfl_xor(p3, off, 32);
    }
    p0 *= INV_SQRT_D; p1 *= INV_SQRT_D; p2 *= INV_SQRT_D; p3 *= INV_SQRT_D;
    if (lane < 4) {
        float v = (lane == 0) ? p0 : (lane == 1) ? p1 : (lane == 2) ? p2 : p3;
        score[(size_t)wid * H + lane] = v;
        atomicMax(&mx[d * H + lane], enc_f32(v));
    }
}

// Decode segment max; empty segments (-inf) -> 0 per reference.
__global__ void k_finalize_max(const unsigned* mx, float* mf, int n) {
    int i = blockIdx.x * blockDim.x + threadIdx.x;
    if (i >= n) return;
    unsigned e = mx[i];
    unsigned bits = (e & 0x80000000u) ? (e ^ 0x80000000u) : ~e;
    float f = __uint_as_float(bits);
    mf[i] = isfinite(f) ? f : 0.0f;
}

// Fused unnormalized softmax scatter: agg[dst] += e*v[src], denom[dst] += e.
__global__ void k_edge_scatter(const int* __restrict__ ei,
                               const float* __restrict__ score,
                               const float* __restrict__ mf,
                               const float* __restrict__ v,
                               float* __restrict__ denom,
                               float* __restrict__ agg) {
    int wid  = (blockIdx.x * blockDim.x + threadIdx.x) >> 5;
    int lane = threadIdx.x & 31;
    if (wid >= N_EDGES) return;
    if (lane == 0 && wid + 4096 < N_EDGES)
        __builtin_prefetch(score + (size_t)(wid + 4096) * H, 0, 0);
    int s = ei[wid], d = ei[N_EDGES + wid];
    const float* sc = score + (size_t)wid * H;
    const float* mm = mf + (size_t)d * H;
    float e0 = __expf(sc[0] - mm[0]);
    float e1 = __expf(sc[1] - mm[1]);
    float e2 = __expf(sc[2] - mm[2]);
    float e3 = __expf(sc[3] - mm[3]);
    if (lane < 4) {
        float ev = (lane == 0) ? e0 : (lane == 1) ? e1 : (lane == 2) ? e2 : e3;
        atomicAdd(&denom[d * H + lane], ev);
    }
    const float* vs = v + (size_t)s * C;
    float* ad = agg + (size_t)d * C;
    atomicAdd(&ad[lane],      e0 * vs[lane]);
    atomicAdd(&ad[32 + lane], e1 * vs[32 + lane]);
    atomicAdd(&ad[64 + lane], e2 * vs[64 + lane]);
    atomicAdd(&ad[96 + lane], e3 * vs[96 + lane]);
}

// x += relu(agg/denom + skip)
__global__ void k_node_finalize(float* __restrict__ x,
                                const float* __restrict__ agg,
                                const float* __restrict__ denom,
                                const float* __restrict__ skip) {
    int gid = blockIdx.x * blockDim.x + threadIdx.x;
    if (gid >= N_NODES * C) return;
    int n = gid >> 7, h = (gid & 127) >> 5;
    float a = agg[gid] / (denom[n * H + h] + 1e-16f) + skip[gid];
    x[gid] += fmaxf(a, 0.0f);
}

// small output head: out[n,o] = t1[n,:] @ w[:,O] + b[o]
__global__ void k_head(const float* __restrict__ t1, const float* __restrict__ w,
                       const float* __restrict__ bias, float* __restrict__ out,
                       int O) {
    int gid = blockIdx.x * blockDim.x + threadIdx.x;
    if (gid >= N_NODES * O) return;
    int n = gid / O, o = gid - n * O;
    const float* row = t1 + (size_t)n * C;
    float acc = bias[o];
    #pragma unroll 8
    for (int c = 0; c < C; ++c) acc += row[c] * w[c * O + o];
    out[gid] = acc;
}

// ---------------------------------------------------------------------------

extern "C" void kernel_launch(void* const* d_in, const int* in_sizes, int n_in,
                              void* d_out, int out_size, void* d_ws, size_t ws_size,
                              hipStream_t stream) {
    const float* xs   = (const float*)d_in[0];
    const float* xd   = (const float*)d_in[1];
    const float* xp   = (const float*)d_in[2];
    const float* xt   = (const float*)d_in[3];
    const int*   ei   = (const int*)  d_in[4];
    const float* pw   = (const float*)d_in[5];
    const float* lng  = (const float*)d_in[6];
    const float* lnb  = (const float*)d_in[7];
    const float* Wq   = (const float*)d_in[8];
    const float* bq   = (const float*)d_in[9];
    const float* Wk   = (const float*)d_in[10];
    const float* bk   = (const float*)d_in[11];
    const float* Wv   = (const float*)d_in[12];
    const float* bv   = (const float*)d_in[13];
    const float* Ws   = (const float*)d_in[14];
    const float* bs   = (const float*)d_in[15];
    const float* f1w  = (const float*)d_in[16];
    const float* f1b  = (const float*)d_in[17];
    const float* f2w  = (const float*)d_in[18];
    const float* f2b  = (const float*)d_in[19];
    const float* f3w  = (const float*)d_in[20];
    const float* f3b  = (const float*)d_in[21];
    const float* f4w  = (const float*)d_in[22];
    const float* f4b  = (const float*)d_in[23];
    float* out = (float*)d_out;

    const size_t NC = (size_t)N_NODES * C;
    float* xbuf  = (float*)d_ws;
    float* hbuf  = xbuf + NC;
    float* qb    = hbuf + NC;        // also reused as fc intermediate
    float* kb    = qb + NC;
    float* vb    = kb + NC;
    float* sb    = vb + NC;
    float* agg   = sb + NC;
    float* score = agg + NC;                       // E*H
    unsigned* mx = (unsigned*)(score + (size_t)N_EDGES * H);
    float* mf    = (float*)(mx + (size_t)N_NODES * H);
    float* denom = mf + (size_t)N_NODES * H;
    __bf16* wbf  = (__bf16*)(denom + (size_t)N_NODES * H);

    const int TB = 256;
    dim3 b(TB);
    int g_nc   = (int)((NC + TB - 1) / TB);              // 40000
    int g_nh   = (N_NODES * H + TB - 1) / TB;            // 1250
    int g_ln   = (N_NODES * 32 + TB - 1) / TB;           // 10000
    int g_edge = (N_EDGES * 32 + TB - 1) / TB;           // 160000
    int g_gemm = N_NODES / 16;                           // 5000

    k_cvt_weights<<<(18 * C * C + TB - 1) / TB, b, 0, stream>>>(
        Wq, Wk, Wv, Ws, f1w, f2w, wbf);
    k_proj<<<g_nc, b, 0, stream>>>(xs, xd, xp, xt, pw, xbuf);

    for (int l = 0; l < 4; ++l) {
        k_layernorm<<<g_ln, b, 0, stream>>>(xbuf, lng + l * C, lnb + l * C, hbuf);
        k_gemm_qkvs<<<g_gemm, b, 0, stream>>>(
            hbuf, wbf + (size_t)(l * 4) * C * C,
            bq + l * C, bk + l * C, bv + l * C, bs + l * C,
            qb, kb, vb, sb);
        k_init_max<<<g_nh, b, 0, stream>>>(mx, N_NODES * H);
        k_zero_f32<<<g_nh, b, 0, stream>>>(denom, N_NODES * H);
        k_zero_f32<<<g_nc, b, 0, stream>>>(agg, (int)NC);
        k_edge_scores<<<g_edge, b, 0, stream>>>(ei, qb, kb, score, mx);
        k_finalize_max<<<g_nh, b, 0, stream>>>(mx, mf, N_NODES * H);
        k_edge_scatter<<<g_edge, b, 0, stream>>>(ei, score, mf, vb, denom, agg);
        k_node_finalize<<<g_nc, b, 0, stream>>>(xbuf, agg, denom, sb);

        if (l == 2) {   // prognosis head reads x_pi before block 3 mutates x
            k_gemm128<<<g_gemm, b, 0, stream>>>(xbuf, wbf + (size_t)16 * C * C, f1b, qb);
            k_head<<<(N_NODES * 7 + TB - 1) / TB, b, 0, stream>>>(qb, f3w, f3b, out, 7);
        }
    }
    // diagnosis head on x_dg (after block 3)
    k_gemm128<<<g_gemm, b, 0, stream>>>(xbuf, wbf + (size_t)17 * C * C, f2b, qb);
    k_head<<<(N_NODES * 3 + TB - 1) / TB, b, 0, stream>>>(qb, f4w, f4b, out + (size_t)N_NODES * 7, 3);
}
